// SambaAttention_63934883168592
// MI455X (gfx1250) — matrix operand order repaired and checked by reading, not compiled, not run on hardware
//
#include <hip/hip_runtime.h>

#define HIDN 4096
#define SLEN 2048
#define WINSZ 1024
#define QK_SCALE 0.08838834764831845f   /* 128^-0.5 */
#define LAMB_INIT 0.35550907f           /* 0.8 - 0.6*exp(-0.3) */
#define TS 20                           /* u32 stride of a 32-bf16 LDS tile row (16 data + 4 pad) */

typedef __bf16 v16bf __attribute__((ext_vector_type(16)));
typedef float v8f __attribute__((ext_vector_type(8)));
typedef unsigned int v8u __attribute__((ext_vector_type(8)));

#define WMMA_BF16(A, B, C) \
  __builtin_amdgcn_wmma_f32_16x16x32_bf16(false, (A), false, (B), (short)0, (C), false, false)

/* Async global->LDS 16B copy (CDNA5; ASYNCcnt tracked). Generic shared
   pointers carry the LDS byte offset in their low 32 bits (aperture form). */
__device__ __forceinline__ void async_cp16(void* lds, const void* g) {
  unsigned int l = (unsigned int)(unsigned long long)lds;
  asm volatile("global_load_async_to_lds_b128 %0, %1, off"
               :: "v"(l), "v"((unsigned long long)g) : "memory");
}
#define WAIT_ASYNC() asm volatile("s_wait_asynccnt 0x0" ::: "memory")

__device__ __forceinline__ v8f vzero() {
  v8f z;
#pragma unroll
  for (int i = 0; i < 8; ++i) z[i] = 0.0f;
  return z;
}

/* f32 -> bf16 packed pair; prefer hw v_cvt_pk_bf16_f32 when available */
#if __has_builtin(__builtin_amdgcn_cvt_pk_bf16_f32)
typedef __bf16 v2bf __attribute__((ext_vector_type(2)));
__device__ __forceinline__ unsigned int pkbf(float lo, float hi) {
  v2bf r = __builtin_amdgcn_cvt_pk_bf16_f32(lo, hi);
  return __builtin_bit_cast(unsigned int, r);
}
__device__ __forceinline__ unsigned short bf1(float x) {
  v2bf r = __builtin_amdgcn_cvt_pk_bf16_f32(x, 0.0f);
  return (unsigned short)(__builtin_bit_cast(unsigned int, r) & 0xffffu);
}
#else
__device__ __forceinline__ unsigned int pkbf(float lo, float hi) {
  unsigned int a = __float_as_uint(lo), b = __float_as_uint(hi);
  a = (a + 0x7fffu + ((a >> 16) & 1u)) >> 16;
  b = (b + 0x7fffu + ((b >> 16) & 1u)) >> 16;
  return a | (b << 16);
}
__device__ __forceinline__ unsigned short bf1(float x) {
  unsigned int a = __float_as_uint(x);
  return (unsigned short)((a + 0x7fffu + ((a >> 16) & 1u)) >> 16);
}
#endif

/* A-operand fragment: 16 rows striped over lane&15; K interleaved halves
   {0..7,16..23} (lanes 0-15) / {8..15,24..31} (lanes 16-31) per ISA A-layout. */
__device__ __forceinline__ v16bf ldfragA(const unsigned int* p, int row0, int stride) {
  const int lane = threadIdx.x & 31;
  const unsigned int* q = p + (row0 + (lane & 15)) * stride + ((lane >> 4) << 2);
  uint4 a = *(const uint4*)q;
  uint4 b = *(const uint4*)(q + 8);
  v8u u;
  u[0] = a.x; u[1] = a.y; u[2] = a.z; u[3] = a.w;
  u[4] = b.x; u[5] = b.y; u[6] = b.z; u[7] = b.w;
  return __builtin_bit_cast(v16bf, u);
}
/* B-operand fragment: 16 columns striped over lane&15; K interleaved halves
   {0..7,16..23} (lanes 0-15) / {8..15,24..31} (lanes 16-31), matching A. */
__device__ __forceinline__ v16bf ldfragB(const unsigned int* p, int row0, int stride) {
  const int lane = threadIdx.x & 31;
  const unsigned int* q = p + (row0 + (lane & 15)) * stride + ((lane >> 4) << 2);
  uint4 a = *(const uint4*)q;
  uint4 b = *(const uint4*)(q + 8);
  v8u u;
  u[0] = a.x; u[1] = a.y; u[2] = a.z; u[3] = a.w;
  u[4] = b.x; u[5] = b.y; u[6] = b.z; u[7] = b.w;
  return __builtin_bit_cast(v16bf, u);
}

/* 256 threads stage a 128x32 f32 tile as bf16 into LDS (row stride TS u32). */
__device__ __forceinline__ void stage_f32_tile(unsigned int* dst, const float* src,
                                               long row_stride, int tid) {
  const int r = tid >> 1, h = tid & 1;
  const float4* p = (const float4*)(src + (long)r * row_stride + h * 16);
  float4 x0 = p[0], x1 = p[1], x2 = p[2], x3 = p[3];
  unsigned int* d = dst + r * TS + h * 8;
  d[0] = pkbf(x0.x, x0.y); d[1] = pkbf(x0.z, x0.w);
  d[2] = pkbf(x1.x, x1.y); d[3] = pkbf(x1.z, x1.w);
  d[4] = pkbf(x2.x, x2.y); d[5] = pkbf(x2.z, x2.w);
  d[6] = pkbf(x3.x, x3.y); d[7] = pkbf(x3.z, x3.w);
}
/* 256 threads stage a 128x32 bf16 tile into LDS via async global->LDS DMA. */
__device__ __forceinline__ void stage_bf16_tile_async(unsigned int* dst, const unsigned int* src,
                                                      long row_u32, int tid) {
  const int r = tid >> 1, h = tid & 1;
  const unsigned int* s = src + (long)r * row_u32 + h * 8;
  unsigned int* d = dst + r * TS + h * 8;
  async_cp16(d, s);
  async_cp16(d + 4, s + 4);
}

/* ------------------------------------------------------------------ */
/* Kernel 1: QKV GEMM (hidden @ Wqkv^T + b) fused with RoPE; emits bf16
   q (pre-scaled by 1/sqrt(D)), k, v.                                   */
/* ------------------------------------------------------------------ */
__global__ __launch_bounds__(256) void k_qkv_rope(
    const float* __restrict__ hid, const float* __restrict__ w,
    const float* __restrict__ bias, const int* __restrict__ pos,
    unsigned short* __restrict__ qb, unsigned short* __restrict__ kb,
    unsigned short* __restrict__ vb) {
  __shared__ unsigned int At[2][128 * TS];
  __shared__ unsigned int Bt[2][128 * TS];
  __shared__ float Ct[32 * 132];
  const int tid = threadIdx.x, lane = tid & 31, wv = tid >> 5;
  const int wm = wv & 3, wn = wv >> 2;
  const int n0 = blockIdx.x * 128, m0 = blockIdx.y * 128;

  v8f acc[2][4];
#pragma unroll
  for (int s = 0; s < 2; ++s)
#pragma unroll
    for (int c = 0; c < 4; ++c) acc[s][c] = vzero();

  stage_f32_tile(At[0], hid + (size_t)m0 * HIDN, HIDN, tid);
  stage_f32_tile(Bt[0], w + (size_t)n0 * HIDN, HIDN, tid);

  const int NK = HIDN / 32;
  for (int ki = 0; ki < NK; ++ki) {
    __syncthreads();
    const int buf = ki & 1;
    if (ki + 1 < NK) {
      stage_f32_tile(At[buf ^ 1], hid + (size_t)m0 * HIDN + (ki + 1) * 32, HIDN, tid);
      stage_f32_tile(Bt[buf ^ 1], w + (size_t)n0 * HIDN + (ki + 1) * 32, HIDN, tid);
    }
    if (ki + 8 < NK) { /* L2 prefetch of future tiles */
      __builtin_prefetch(hid + (size_t)(m0 + (tid >> 1)) * HIDN + (ki + 8) * 32 + (tid & 1) * 16, 0, 0);
      __builtin_prefetch(w + (size_t)(n0 + (tid >> 1)) * HIDN + (ki + 8) * 32 + (tid & 1) * 16, 0, 0);
    }
    v16bf a0 = ldfragA(At[buf], wm * 32, TS);
    v16bf a1 = ldfragA(At[buf], wm * 32 + 16, TS);
#pragma unroll
    for (int ct = 0; ct < 4; ++ct) {
      v16bf bb = ldfragB(Bt[buf], wn * 64 + ct * 16, TS);
      acc[0][ct] = WMMA_BF16(a0, bb, acc[0][ct]);
      acc[1][ct] = WMMA_BF16(a1, bb, acc[1][ct]);
    }
  }

  /* Epilogue: bounce C through LDS 32 rows at a time; bias + RoPE + bf16. */
  for (int rc = 0; rc < 4; ++rc) {
    __syncthreads();
    if (wm == rc) {
#pragma unroll
      for (int sub = 0; sub < 2; ++sub)
#pragma unroll
        for (int ct = 0; ct < 4; ++ct)
#pragma unroll
          for (int e = 0; e < 8; ++e) {
            int lr = sub * 16 + e + ((lane >> 4) << 3);
            int cc = wn * 64 + ct * 16 + (lane & 15);
            Ct[lr * 132 + cc] = acc[sub][ct][e];
          }
    }
    __syncthreads();
    const int rbase = m0 + rc * 32;
    if (n0 < 5120) { /* q or k head (one head per 128-col tile): RoPE */
      const bool isq = (n0 < HIDN);
      for (int it = tid; it < 32 * 64; it += 256) {
        int r = it >> 6, d = it & 63;
        int grow = rbase + r;
        float x1 = Ct[r * 132 + d] + bias[n0 + d];
        float x2 = Ct[r * 132 + d + 64] + bias[n0 + d + 64];
        float fr = __powf(10000.0f, -(float)d * (1.0f / 64.0f));
        float sn, cs;
        __sincosf((float)pos[grow] * fr, &sn, &cs);
        float o1 = x1 * cs - x2 * sn;
        float o2 = x2 * cs + x1 * sn;
        if (isq) {
          o1 *= QK_SCALE; o2 *= QK_SCALE;
          qb[(size_t)grow * HIDN + n0 + d] = bf1(o1);
          qb[(size_t)grow * HIDN + n0 + d + 64] = bf1(o2);
        } else {
          int c0 = n0 - HIDN;
          kb[(size_t)grow * 1024 + c0 + d] = bf1(o1);
          kb[(size_t)grow * 1024 + c0 + d + 64] = bf1(o2);
        }
      }
    } else { /* v head: bias + convert */
      int c0 = n0 - 5120;
      for (int it = tid; it < 32 * 128; it += 256) {
        int r = it >> 7, d = it & 127;
        int grow = rbase + r;
        vb[(size_t)grow * 1024 + c0 + d] = bf1(Ct[r * 132 + d] + bias[n0 + d]);
      }
    }
  }
}

/* ------------------------------------------------------------------ */
/* Kernel 2: flash-style differential attention with sliding window.   */
/* One WG = 32 query rows x 1 diff-head; 4 waves (16 rows x 128 cols). */
/* ------------------------------------------------------------------ */
__global__ __launch_bounds__(128) void k_diffattn(
    const unsigned short* __restrict__ qb, const unsigned short* __restrict__ kb,
    const unsigned short* __restrict__ vb,
    const float* __restrict__ lq1, const float* __restrict__ lk1,
    const float* __restrict__ lq2, const float* __restrict__ lk2,
    const float* __restrict__ subw, unsigned short* __restrict__ attn) {
  __shared__ unsigned int Qt[32 * 132];     /* [row][q1(128)|q2(128)] bf16 */
  __shared__ unsigned int Kt[2][32 * 68];   /* [key][dim] bf16, per branch */
  __shared__ unsigned int Vt[2][128 * TS];  /* [dim][key] bf16 (transposed) */
  __shared__ unsigned int Pt[4][16 * TS];   /* per-wave prob scratch */
  __shared__ float sums[32][2];
  __shared__ float red1[128], red2[128];
  __shared__ float s_lam;

  const int tid = threadIdx.x, lane = tid & 31, wv = tid >> 5;
  const int rg = wv >> 1, hf = wv & 1;
  const int qb0 = blockIdx.x * 32;
  const int head = blockIdx.y;       /* diff-head 0..15 */
  const int kvh = 2 * (head >> 2);   /* kv head pair base */

  red1[tid] = lq1[tid] * lk1[tid];
  red2[tid] = lq2[tid] * lk2[tid];

  { /* stage Q via async DMA: both sub-heads contiguous 256 bf16 per row */
    int r = tid >> 2, part = tid & 3;
    const unsigned int* src =
        (const unsigned int*)qb + (size_t)(qb0 + r) * (HIDN / 2) + head * 128 + part * 32;
    unsigned int* dst = Qt + r * 132 + part * 32;
#pragma unroll
    for (int q8 = 0; q8 < 8; ++q8) async_cp16(dst + q8 * 4, src + q8 * 4);
  }
  WAIT_ASYNC();
  __syncthreads();
  if (tid == 0) {
    float a = 0.f, b = 0.f;
    for (int j = 0; j < 128; ++j) { a += red1[j]; b += red2[j]; }
    s_lam = __expf(a) - __expf(b) + LAMB_INIT;
  }

  v8f O1[8], O2[8];
#pragma unroll
  for (int c = 0; c < 8; ++c) { O1[c] = vzero(); O2[c] = vzero(); }
  float m1[8], l1[8], m2[8], l2[8];
#pragma unroll
  for (int e = 0; e < 8; ++e) { m1[e] = -1e30f; m2[e] = -1e30f; l1[e] = 0.f; l2[e] = 0.f; }

  int kstart = qb0 - (WINSZ - 1);
  if (kstart < 0) kstart = 0;
  kstart &= ~31;

  auto do_branch = [&](const unsigned int* ktile, int qofs, v8f (&O)[8],
                       float (&m)[8], float (&l)[8], int kk) {
    v8f S0 = vzero(), S1 = vzero();
#pragma unroll
    for (int ks = 0; ks < 4; ++ks) {
      v16bf qa = ldfragA(Qt + qofs + ks * 16, rg * 16, 132);
      v16bf k0 = ldfragB(ktile + ks * 16, 0, 68);
      v16bf k1 = ldfragB(ktile + ks * 16, 16, 68);
      S0 = WMMA_BF16(qa, k0, S0);
      S1 = WMMA_BF16(qa, k1, S1);
    }
    /* sliding-window causal mask */
    float s0[8], s1[8];
    const int cb = kk + (lane & 15);
#pragma unroll
    for (int e = 0; e < 8; ++e) {
      int grow = qb0 + rg * 16 + e + ((lane >> 4) << 3);
      s0[e] = (cb <= grow && grow - cb < WINSZ) ? S0[e] : -1e30f;
      int c1 = cb + 16;
      s1[e] = (c1 <= grow && grow - c1 < WINSZ) ? S1[e] : -1e30f;
    }
    /* per-row online max (rows live in 16-lane halves of wave32) */
    float fac[8];
#pragma unroll
    for (int e = 0; e < 8; ++e) {
      float mb = fmaxf(s0[e], s1[e]);
#pragma unroll
      for (int off = 1; off < 16; off <<= 1) mb = fmaxf(mb, __shfl_xor(mb, off, 32));
      float mn = fmaxf(m[e], mb);
      fac[e] = __expf(m[e] - mn);
      m[e] = mn;
    }
    unsigned short* pw = (unsigned short*)Pt[wv];
#pragma unroll
    for (int e = 0; e < 8; ++e) {
      float p0 = (s0[e] <= -1e29f) ? 0.f : __expf(s0[e] - m[e]);
      float p1 = (s1[e] <= -1e29f) ? 0.f : __expf(s1[e] - m[e]);
      float rs = p0 + p1;
#pragma unroll
      for (int off = 1; off < 16; off <<= 1) rs += __shfl_xor(rs, off, 32);
      l[e] = l[e] * fac[e] + rs;
      int rr = e + ((lane >> 4) << 3);
      pw[rr * (TS * 2) + (lane & 15)] = bf1(p0);
      pw[rr * (TS * 2) + 16 + (lane & 15)] = bf1(p1);
    }
    asm volatile("" ::: "memory"); /* keep LDS prob stores before frag reload */
    v16bf pf = ldfragA(Pt[wv], 0, TS);
#pragma unroll
    for (int ct = 0; ct < 8; ++ct) {
#pragma unroll
      for (int e = 0; e < 8; ++e) O[ct][e] *= fac[e];
      v16bf vf = ldfragB(Vt[hf], ct * 16, TS);
      O[ct] = WMMA_BF16(pf, vf, O[ct]);
    }
  };

  for (int kk = kstart; kk <= qb0; kk += 32) {
    __syncthreads();
    { /* stage K1/K2 as [key][dim] via async DMA */
      int h = tid >> 6, tt = tid & 63;
      int j = tt >> 1, pp = tt & 1;
      const unsigned int* src =
          (const unsigned int*)kb + (size_t)(kk + j) * 512 + (kvh + h) * 64 + pp * 32;
      unsigned int* dst = Kt[h] + j * 68 + pp * 32;
#pragma unroll
      for (int q8 = 0; q8 < 8; ++q8) async_cp16(dst + q8 * 4, src + q8 * 4);
    }
    { /* stage V1/V2 transposed as [dim][key] (overlaps with async K) */
      int h = tid >> 6, tt = tid & 63;
      int j = tt & 31, dh = tt >> 5;
      const unsigned int* src =
          (const unsigned int*)vb + (size_t)(kk + j) * 512 + (kvh + h) * 64 + dh * 32;
      unsigned short* dst = (unsigned short*)Vt[h];
#pragma unroll
      for (int q8 = 0; q8 < 8; ++q8) {
        uint4 x = ((const uint4*)src)[q8];
        unsigned int xs[4] = {x.x, x.y, x.z, x.w};
#pragma unroll
        for (int e = 0; e < 4; ++e) {
          int d0 = dh * 64 + q8 * 8 + e * 2;
          dst[d0 * (TS * 2) + j] = (unsigned short)(xs[e] & 0xffffu);
          dst[(d0 + 1) * (TS * 2) + j] = (unsigned short)(xs[e] >> 16);
        }
      }
    }
    WAIT_ASYNC();
    __syncthreads();
    do_branch(Kt[0], 0, O1, m1, l1, kk);
    do_branch(Kt[1], 64, O2, m2, l2, kk);
  }

  /* Epilogue: normalize both branches, combine, RMS-norm over 256 dims. */
  const float lam = s_lam;
  float ssq[8], i1[8], i2[8];
#pragma unroll
  for (int e = 0; e < 8; ++e) {
    ssq[e] = 0.f;
    i1[e] = 1.0f / l1[e];
    i2[e] = 1.0f / l2[e];
  }
#pragma unroll
  for (int ct = 0; ct < 8; ++ct)
#pragma unroll
    for (int e = 0; e < 8; ++e) {
      float v = O1[ct][e] * i1[e] - lam * (O2[ct][e] * i2[e]);
      O1[ct][e] = v;
      ssq[e] += v * v;
    }
#pragma unroll
  for (int e = 0; e < 8; ++e) {
    float t = ssq[e];
#pragma unroll
    for (int off = 1; off < 16; off <<= 1) t += __shfl_xor(t, off, 32);
    ssq[e] = t;
  }
  __syncthreads();
  if ((lane & 15) == 0) {
    int rb = rg * 16 + ((lane >> 4) << 3);
#pragma unroll
    for (int e = 0; e < 8; ++e) sums[rb + e][hf] = ssq[e];
  }
  __syncthreads();
#pragma unroll
  for (int ct = 0; ct < 8; ++ct) {
    int cloc = hf * 128 + ct * 16 + (lane & 15);
    float sw = subw[cloc] * (1.0f - LAMB_INIT);
#pragma unroll
    for (int e = 0; e < 8; ++e) {
      int rl = rg * 16 + e + ((lane >> 4) << 3);
      float tot = sums[rl][0] + sums[rl][1];
      float rms = rsqrtf(tot * (1.0f / 256.0f) + 1e-5f);
      int grow = qb0 + rl;
      attn[(size_t)grow * HIDN + head * 256 + cloc] = bf1(O1[ct][e] * rms * sw);
    }
  }
}

/* ------------------------------------------------------------------ */
/* Kernel 3: output projection attn(bf16) @ out_w^T + out_b -> f32.    */
/* A tile staged by async global->LDS DMA; B tile converted f32->bf16. */
/* ------------------------------------------------------------------ */
__global__ __launch_bounds__(256) void k_outproj(
    const unsigned short* __restrict__ attn, const float* __restrict__ w,
    const float* __restrict__ bias, float* __restrict__ out) {
  __shared__ unsigned int At[2][128 * TS];
  __shared__ unsigned int Bt[2][128 * TS];
  const int tid = threadIdx.x, lane = tid & 31, wv = tid >> 5;
  const int wm = wv & 3, wn = wv >> 2;
  const int n0 = blockIdx.x * 128, m0 = blockIdx.y * 128;

  v8f acc[2][4];
#pragma unroll
  for (int s = 0; s < 2; ++s)
#pragma unroll
    for (int c = 0; c < 4; ++c) acc[s][c] = vzero();

  stage_bf16_tile_async(At[0], (const unsigned int*)attn + (size_t)m0 * 2048, 2048, tid);
  stage_f32_tile(Bt[0], w + (size_t)n0 * HIDN, HIDN, tid);

  const int NK = HIDN / 32;
  for (int ki = 0; ki < NK; ++ki) {
    WAIT_ASYNC();
    __syncthreads();
    const int buf = ki & 1;
    if (ki + 1 < NK) {
      stage_bf16_tile_async(At[buf ^ 1],
                            (const unsigned int*)attn + (size_t)m0 * 2048 + (ki + 1) * 16,
                            2048, tid);
      stage_f32_tile(Bt[buf ^ 1], w + (size_t)n0 * HIDN + (ki + 1) * 32, HIDN, tid);
    }
    if (ki + 8 < NK)
      __builtin_prefetch(w + (size_t)(n0 + (tid >> 1)) * HIDN + (ki + 8) * 32 + (tid & 1) * 16, 0, 0);
    v16bf a0 = ldfragA(At[buf], wm * 32, TS);
    v16bf a1 = ldfragA(At[buf], wm * 32 + 16, TS);
#pragma unroll
    for (int ct = 0; ct < 4; ++ct) {
      v16bf bb = ldfragB(Bt[buf], wn * 64 + ct * 16, TS);
      acc[0][ct] = WMMA_BF16(a0, bb, acc[0][ct]);
      acc[1][ct] = WMMA_BF16(a1, bb, acc[1][ct]);
    }
  }
#pragma unroll
  for (int sub = 0; sub < 2; ++sub)
#pragma unroll
    for (int ct = 0; ct < 4; ++ct) {
      int col = n0 + wn * 64 + ct * 16 + (lane & 15);
      float bbv = bias[col];
#pragma unroll
      for (int e = 0; e < 8; ++e) {
        int row = m0 + wm * 32 + sub * 16 + e + ((lane >> 4) << 3);
        out[(size_t)row * HIDN + col] = acc[sub][ct][e] + bbv;
      }
    }
}

extern "C" void kernel_launch(void* const* d_in, const int* in_sizes, int n_in,
                              void* d_out, int out_size, void* d_ws, size_t ws_size,
                              hipStream_t stream) {
  (void)in_sizes; (void)n_in; (void)out_size; (void)ws_size;
  const float* hid = (const float*)d_in[0];
  const int* pos = (const int*)d_in[1];
  const float* wqkv = (const float*)d_in[2];
  const float* bqkv = (const float*)d_in[3];
  const float* outw = (const float*)d_in[4];
  const float* outb = (const float*)d_in[5];
  const float* lq1 = (const float*)d_in[6];
  const float* lk1 = (const float*)d_in[7];
  const float* lq2 = (const float*)d_in[8];
  const float* lk2 = (const float*)d_in[9];
  const float* subw = (const float*)d_in[10];

  char* ws = (char*)d_ws;
  unsigned short* qb = (unsigned short*)ws;   ws += (size_t)SLEN * HIDN * 2;  /* 16 MB */
  unsigned short* kb = (unsigned short*)ws;   ws += (size_t)SLEN * 1024 * 2;  /*  4 MB */
  unsigned short* vb = (unsigned short*)ws;   ws += (size_t)SLEN * 1024 * 2;  /*  4 MB */
  unsigned short* attn = (unsigned short*)ws;                                  /* 16 MB */

  k_qkv_rope<<<dim3(48, 16), 256, 0, stream>>>(hid, wqkv, bqkv, pos, qb, kb, vb);
  k_diffattn<<<dim3(64, 16), 128, 0, stream>>>(qb, kb, vb, lq1, lk1, lq2, lk2, subw, attn);
  k_outproj<<<dim3(32, 16), 256, 0, stream>>>(attn, outw, outb, (float*)d_out);
}
